// TransformerEncoder_50087908606509
// MI455X (gfx1250) — matrix operand order repaired and checked by reading, not compiled, not run on hardware
//
#include <hip/hip_runtime.h>

// ---------------------------------------------------------------------------
// CDNA5 (gfx1250) fused linear-attention transformer stack.
// wave32, V_WMMA_F32_16X16X32_BF16. Two sequences per workgroup (128 residual
// rows in LDS); each wave owns two 16-row M-tiles so every B fragment loaded
// from L2 is used twice. k-outer loop: each A fragment loaded from LDS once,
// feeding 16 independent WMMAs.
// ---------------------------------------------------------------------------

typedef __attribute__((ext_vector_type(16))) __bf16 v16bf;
typedef __attribute__((ext_vector_type(8)))  float  v8f;

#define TDIM     128     // model dim
#define THEADS   8
#define TDH      16      // dim per head
#define TFF      512     // FF dim
#define TSP      64      // padded sequence length (per sequence)
#define TROWS    128     // 2 sequences * 64 padded rows
#define TEPS     1e-5f
#define NTHREADS 128     // 4 waves

// ---- LDS layout (bytes), total 304 KB ----
#define OFF_HB   0                        // float  [128][128] residual (2 seqs)
#define OFF_AB   (OFF_HB + 128*128*4)     // ushort [128][128] bf16 A operand
#define OFF_MM   (OFF_AB + 128*128*2)     // float  [128][128] matmul staging
#define OFF_Q2   (OFF_MM + 128*128*4)     // ushort [128][256] q head-padded
#define OFF_KT   (OFF_Q2 + 128*256*2)     // ushort [2][128][64] k transposed
#define OFF_VF   (OFF_KT + 2*128*64*2)    // ushort [2][2][8][32][16] v B-frag
#define OFF_CX   (OFF_VF + 2*8192*2)      // ushort [2][8][32][16] ctx B-frag
#define OFF_FF   OFF_Q2                   // ushort [128][512] (aliases Q2/KT/VF)
#define LDS_BYTES (OFF_CX + 2*4096*2)     // 311296 bytes = 304 KB

// bf16 swizzled blob layout per stack (element offsets, 4 layers each)
#define BW_Q   0
#define BW_K   65536
#define BW_V   131072
#define BW_O   196608
#define BW_1   262144
#define BW_2   524288
#define BW_STACK 786432

static __device__ __forceinline__ unsigned short f2bf(float f) {
  union { float f; unsigned u; } c; c.f = f;
  unsigned r = c.u + 0x7FFFu + ((c.u >> 16) & 1u);   // round-to-nearest-even
  return (unsigned short)(r >> 16);
}

union BFrag { v16bf v; unsigned short s[16]; uint4 q[2]; };

static __device__ __forceinline__ v8f wmma_bf16(v16bf a, v16bf b, v8f c) {
  return __builtin_amdgcn_wmma_f32_16x16x32_bf16(
      /*neg_a=*/false, a, /*neg_b=*/false, b,
      /*c_mod=*/(short)0, c, /*reuse_a=*/false, /*reuse_b=*/false);
}

// A fragment from row-major LDS [..][lda] bf16; 16x32 tile at (m0,k0).
// Lane l<16: row m0+l, K k0+{0..15}; lanes 16-31: K k0+{16..31}.
static __device__ __forceinline__ v16bf load_a(const unsigned short* A, int lda,
                                               int m0, int k0, int lane) {
  BFrag f;
  const int m  = m0 + (lane & 15);
  const int kb = k0 + ((lane & 16) ? 16 : 0);
  const uint4* p = (const uint4*)(A + m * lda + kb);
  f.q[0] = p[0];          // halves [0..7]
  f.q[1] = p[1];          // halves [8..15]
  return f.v;
}

// B fragment from swizzled blob: frag = 32 lanes x 16 halves contiguous.
static __device__ __forceinline__ v16bf load_bswz(const unsigned short* B,
                                                  int fragIdx, int lane) {
  BFrag f;
  const uint4* p = (const uint4*)(B + (fragIdx * 32 + lane) * 16);
  f.q[0] = p[0];
  f.q[1] = p[1];
  return f.v;
}

// C/D 16x16 f32: lanes 0-15 -> rows m0+vgpr, col n0+l; lanes 16-31 -> +8 rows.
static __device__ __forceinline__ void store_c(float* C, int ldc, int m0, int n0,
                                               int lane, v8f acc) {
  const int n  = n0 + (lane & 15);
  const int mb = m0 + ((lane & 16) ? 8 : 0);
#pragma unroll
  for (int j = 0; j < 8; ++j) C[(mb + j) * ldc + n] = acc[j];
}

// [128 x K] @ [K x 128-col-slice] -> fp32 LDS [128][128].
// Wave w owns M-tiles at rows 16w (seq 0) and 64+16w (seq 1); each B fragment
// is loaded once and used by both tiles. k-outer: A fragments loaded once.
static __device__ __forceinline__ void mm128(const unsigned short* __restrict__ A, int K,
                                             const unsigned short* __restrict__ Bswz,
                                             int ntiles, int nt_base,
                                             float* __restrict__ C) {
  const int wave = threadIdx.x >> 5;
  const int lane = threadIdx.x & 31;
  const int m0 = wave * 16;
  const int m1 = 64 + wave * 16;
  v8f acc0[8] = {};
  v8f acc1[8] = {};
  for (int k0 = 0; k0 < K; k0 += 32) {
    const v16bf a0 = load_a(A, K, m0, k0, lane);
    const v16bf a1 = load_a(A, K, m1, k0, lane);
    const int fbase = (k0 >> 5) * ntiles + nt_base;
#pragma unroll
    for (int nt = 0; nt < 8; ++nt) {
      const v16bf bv = load_bswz(Bswz, fbase + nt, lane);
      acc0[nt] = wmma_bf16(a0, bv, acc0[nt]);
      acc1[nt] = wmma_bf16(a1, bv, acc1[nt]);
    }
  }
#pragma unroll
  for (int nt = 0; nt < 8; ++nt) {
    store_c(C, TDIM, m0, nt * 16, lane, acc0[nt]);
    store_c(C, TDIM, m1, nt * 16, lane, acc1[nt]);
  }
  __syncthreads();
}

static __device__ __forceinline__ float gelu_tanh(float x) {
  const float k0 = 0.7978845608028654f;   // sqrt(2/pi)
  return 0.5f * x * (1.0f + tanhf(k0 * (x + 0.044715f * x * x * x)));
}

// LayerNorm all 128 rows of H into bf16 A operand; rows with (r&63)>=S zeroed.
static __device__ __forceinline__ void layernorm_rows(const float* __restrict__ H,
                                                      unsigned short* __restrict__ A,
                                                      const float* __restrict__ g,
                                                      const float* __restrict__ b, int S) {
  const int t = threadIdx.x;            // one row per thread, 128 rows
  unsigned short* ar = A + t * TDIM;
  if ((t & 63) < S) {
    const float* row = H + t * TDIM;
    float m = 0.f;
    for (int e = 0; e < TDIM; ++e) m += row[e];
    m *= (1.f / TDIM);
    float v = 0.f;
    for (int e = 0; e < TDIM; ++e) { float d = row[e] - m; v += d * d; }
    v *= (1.f / TDIM);
    const float rs = rsqrtf(v + TEPS);
    for (int e = 0; e < TDIM; ++e)
      ar[e] = f2bf((row[e] - m) * rs * g[e] + b[e]);
  } else {
    for (int e = 0; e < TDIM; ++e) ar[e] = 0;
  }
  __syncthreads();
}

// ---------------------------------------------------------------------------
// Fused 4-layer stack; TWO sequences per workgroup (rows 0-63 / 64-127).
// mode 0: gather from x[B=64, E=128, C=62, P=64], seq = b*64+p, S=62.
// mode 1: row-major input [nseq, 64, 128], S=64.
// Writes per-sequence token means -> out[seq*128 + e].
// ---------------------------------------------------------------------------
__global__ __launch_bounds__(NTHREADS)
void stack_kernel(const float* __restrict__ in, int mode, int S,
                  const unsigned short* __restrict__ blob,
                  const float* __restrict__ bo,  const float* __restrict__ g1,
                  const float* __restrict__ be1, const float* __restrict__ g2,
                  const float* __restrict__ be2, const float* __restrict__ bf1,
                  const float* __restrict__ bf2, float* __restrict__ out) {
  extern __shared__ char smem[];
  float*          HB  = (float*)(smem + OFF_HB);             // [128][128]
  unsigned short* AB  = (unsigned short*)(smem + OFF_AB);    // [128][128]
  float*          MM  = (float*)(smem + OFF_MM);             // [128][128]
  unsigned short* Q2  = (unsigned short*)(smem + OFF_Q2);    // [128][256]
  unsigned short* KT  = (unsigned short*)(smem + OFF_KT);    // [2][128][64]
  unsigned short* VF  = (unsigned short*)(smem + OFF_VF);    // [2] frag layout
  unsigned short* CX  = (unsigned short*)(smem + OFF_CX);    // [2] frag layout
  unsigned short* FFB = (unsigned short*)(smem + OFF_FF);    // [128][512]

  const int t = threadIdx.x;
  const int wave = t >> 5, lane = t & 31;

  // ---- load two sequence tiles ----
  for (int half = 0; half < 2; ++half) {
    const int s = 2 * blockIdx.x + half;
    float* hb = HB + half * TSP * TDIM;
    if (mode == 0) {
      const int b = s >> 6, p = s & 63;
      for (int idx = t; idx < 62 * TDIM; idx += NTHREADS) {
        const int c = idx >> 7, e = idx & 127;
        hb[idx] = in[((b * TDIM + e) * 62 + c) * 64 + p];
      }
      for (int idx = 62 * TDIM + t; idx < TSP * TDIM; idx += NTHREADS) hb[idx] = 0.f;
    } else {
      for (int idx = t; idx < TSP * TDIM; idx += NTHREADS)
        hb[idx] = in[s * TSP * TDIM + idx];
    }
  }
  __syncthreads();

  for (int l = 0; l < 4; ++l) {
    const unsigned short* wq = blob + BW_Q + l * 16384;
    const unsigned short* wk = blob + BW_K + l * 16384;
    const unsigned short* wv = blob + BW_V + l * 16384;
    const unsigned short* wo = blob + BW_O + l * 16384;
    const unsigned short* w1 = blob + BW_1 + l * 65536;
    const unsigned short* w2 = blob + BW_2 + l * 65536;
    const float* bo_l  = bo  + l * TDIM;
    const float* g1_l  = g1  + l * TDIM;
    const float* be1_l = be1 + l * TDIM;
    const float* g2_l  = g2  + l * TDIM;
    const float* be2_l = be2 + l * TDIM;
    const float* bf1_l = bf1 + l * TFF;
    const float* bf2_l = bf2 + l * TDIM;

    // ================= linear attention =================
    layernorm_rows(HB, AB, g1_l, be1_l, S);

    // q = ln @ Wq ; softmax over each head's 16 features; * dh^-0.5.
    // Head-padded: Q2[r][h*32 + i], cols h*32+16..31 zero (K padding).
    mm128(AB, TDIM, wq, 8, 0, MM);
    {
      unsigned short* qr = Q2 + t * 256;
      if ((t & 63) < S) {
        const float* r = MM + t * TDIM;
        for (int h = 0; h < THEADS; ++h) {
          float mx = -1e30f;
          for (int i = 0; i < TDH; ++i) mx = fmaxf(mx, r[h * TDH + i]);
          float sum = 0.f;
          for (int i = 0; i < TDH; ++i) sum += __expf(r[h * TDH + i] - mx);
          const float inv = 0.25f / sum;   // * dh^-0.5
          for (int i = 0; i < TDH; ++i) {
            qr[h * 32 + i]      = f2bf(__expf(r[h * TDH + i] - mx) * inv);
            qr[h * 32 + 16 + i] = 0;
          }
        }
      } else {
        for (int e = 0; e < 256; ++e) qr[e] = 0;
      }
    }
    __syncthreads();

    // k = ln @ Wk ; softmax over sequence per (head,feature) column, per seq.
    // Stored transposed: KT[seq][feature][n] -> contiguous ctx A-fragments.
    mm128(AB, TDIM, wk, 8, 0, MM);
    for (int half = 0; half < 2; ++half) {
      const float* Mh = MM + half * TSP * TDIM;
      unsigned short* kr = KT + half * 8192 + t * TSP;
      float mx = -1e30f;
      for (int n = 0; n < S; ++n) mx = fmaxf(mx, Mh[n * TDIM + t]);
      float sum = 0.f;
      for (int n = 0; n < S; ++n) sum += __expf(Mh[n * TDIM + t] - mx);
      const float inv = 1.f / sum;
      for (int n = 0; n < S; ++n)
        kr[n] = f2bf(__expf(Mh[n * TDIM + t] - mx) * inv);
      for (int n = S; n < TSP; ++n) kr[n] = 0;
    }
    __syncthreads();

    // v = ln @ Wv ; stored per-seq in B-fragment layout:
    // VF[seq][((kt*8+h)*32+lane)*16+j] = v[kt*32+((lane&16)?16:0)+j][h*16+(lane&15)]
    mm128(AB, TDIM, wv, 8, 0, MM);
    for (int q = 0; q < 128; ++q) {
      const int idx  = q * NTHREADS + t;        // 0 .. 16383
      const int half = idx >> 13;
      const int r    = idx & 8191;
      const int j    = r & 15;
      const int ln   = (r >> 4) & 31;
      const int nt   = (r >> 9) & 7;
      const int kt   = r >> 12;
      const int n    = kt * 32 + ((ln & 16) ? 16 : 0) + j;
      const int col  = nt * 16 + (ln & 15);
      VF[idx] = (n < S) ? f2bf(MM[(half * TSP + n) * TDIM + col]) : (unsigned short)0;
    }
    for (int i = t; i < 2 * 4096; i += NTHREADS) CX[i] = 0;
    __syncthreads();

    // ctx[seq][h] = k_h^T @ v_h (16x16); stored into CX in B-frag layout with
    // K rows 16..31 zero. 16 (seq,head) tasks over 4 waves.
    for (int i = 0; i < 4; ++i) {
      const int pidx = wave * 4 + i;
      const int half = pidx >> 3, h = pidx & 7, hb = h * TDH;
      const unsigned short* kth = KT + half * 8192;
      const unsigned short* vfh = VF + half * 8192;
      v8f acc = {};
      for (int kt = 0; kt < 2; ++kt) {
        v16bf av = load_a(kth, TSP, hb, kt * 32, lane);    // A[i][n] = KT[hb+i][n]
        v16bf bv = load_bswz(vfh, kt * 8 + h, lane);
        acc = wmma_bf16(av, bv, acc);
      }
      const int j  = lane & 15;
      const int mb = (lane & 16) ? 8 : 0;
      unsigned short* cxh = CX + half * 4096;
#pragma unroll
      for (int r = 0; r < 8; ++r)
        cxh[(h * 32 + j) * 16 + (mb + r)] = f2bf(acc[r]);  // frag[lane=j][K=mb+r]
    }
    __syncthreads();

    // o_h = q_h @ ctx_h : one K=32 WMMA per head per M-tile.
    for (int half = 0; half < 2; ++half) {
      const int m0 = half * TSP + wave * 16;
      const unsigned short* cxh = CX + half * 4096;
      for (int h = 0; h < THEADS; ++h) {
        v16bf av = load_a(Q2, 256, m0, h * 32, lane);
        v16bf bv = load_bswz(cxh, h, lane);
        v8f acc = {};
        acc = wmma_bf16(av, bv, acc);
        store_c(MM, TDIM, m0, h * TDH, lane, acc);
      }
    }
    __syncthreads();
    for (int n = 0; n < TROWS; ++n)
      AB[n * TDIM + t] = ((n & 63) < S) ? f2bf(MM[n * TDIM + t]) : (unsigned short)0;
    __syncthreads();

    // attn_out = o @ Wo + bo ; residual
    mm128(AB, TDIM, wo, 8, 0, MM);
    for (int n = 0; n < TROWS; ++n)
      if ((n & 63) < S) HB[n * TDIM + t] += MM[n * TDIM + t] + bo_l[t];
    __syncthreads();

    // ================= feed-forward =================
    layernorm_rows(HB, AB, g2_l, be2_l, S);
    for (int c = 0; c < 4; ++c) {            // FF1 in 128-col chunks
      mm128(AB, TDIM, w1, 32, c * 8, MM);
      for (int n = 0; n < TROWS; ++n) {
        const float v = MM[n * TDIM + t] + bf1_l[c * TDIM + t];
        FFB[n * TFF + c * TDIM + t] =
            ((n & 63) < S) ? f2bf(gelu_tanh(v)) : (unsigned short)0;
      }
      __syncthreads();
    }
    mm128(FFB, TFF, w2, 8, 0, MM);           // FF2, K = 512
    for (int n = 0; n < TROWS; ++n)
      if ((n & 63) < S) HB[n * TDIM + t] += MM[n * TDIM + t] + bf2_l[t];
    __syncthreads();
  }

  // ---- per-sequence mean over tokens ----
  for (int half = 0; half < 2; ++half) {
    float acc = 0.f;
    for (int n = 0; n < S; ++n) acc += HB[(half * TSP + n) * TDIM + t];
    out[(2 * blockIdx.x + half) * TDIM + t] = acc / (float)S;
  }
}

// fp32 [L][K][N] -> bf16 blob in B-fragment order:
// dst[((kt*(N/16)+nt)*32 + lane)*16 + j] = src[k][n],
//   k = kt*32 + ((lane&16)?16:0) + j,  n = nt*16 + (lane&15), per layer.
__global__ void cvt_swz(const float* __restrict__ src,
                        unsigned short* __restrict__ dst,
                        int L, int K, int N) {
  const int idx = blockIdx.x * blockDim.x + threadIdx.x;
  const int kn = K * N;
  if (idx >= L * kn) return;
  const int layer = idx / kn;
  const int r = idx - layer * kn;
  const int j    = r & 15;
  const int lane = (r >> 4) & 31;
  const int frag = r >> 9;
  const int ntiles = N >> 4;
  const int kt = frag / ntiles;
  const int nt = frag - kt * ntiles;
  const int k = kt * 32 + ((lane & 16) ? 16 : 0) + j;
  const int n = nt * 16 + (lane & 15);
  dst[idx] = f2bf(src[layer * kn + k * N + n]);
}

// ---------------------------------------------------------------------------
extern "C" void kernel_launch(void* const* d_in, const int* in_sizes, int n_in,
                              void* d_out, int out_size, void* d_ws, size_t ws_size,
                              hipStream_t stream) {
  (void)in_sizes; (void)n_in; (void)out_size; (void)ws_size;

  const float* x = (const float*)d_in[0];
  // input order: x, then {Wq,Wk,Wv,Wo,bo,g1,be1,g2,be2,W1,bf1,W2,bf2}_c, then _t
  unsigned short* blob_c = (unsigned short*)d_ws;
  unsigned short* blob_t = blob_c + BW_STACK;
  float* stageA = (float*)((char*)d_ws + (4u << 20));  // 4096*128 fp32

  const int w_src_c[6] = {1, 2, 3, 4, 10, 12};
  const int w_src_t[6] = {14, 15, 16, 17, 23, 25};
  const int w_off[6]   = {BW_Q, BW_K, BW_V, BW_O, BW_1, BW_2};
  const int w_K[6]     = {128, 128, 128, 128, 128, 512};
  const int w_N[6]     = {128, 128, 128, 128, 512, 128};
  for (int i = 0; i < 6; ++i) {
    const int n = 4 * w_K[i] * w_N[i];
    cvt_swz<<<(n + 255) / 256, 256, 0, stream>>>(
        (const float*)d_in[w_src_c[i]], blob_c + w_off[i], 4, w_K[i], w_N[i]);
    cvt_swz<<<(n + 255) / 256, 256, 0, stream>>>(
        (const float*)d_in[w_src_t[i]], blob_t + w_off[i], 4, w_K[i], w_N[i]);
  }

  // Stage A: channel attention, 4096 sequences of 62 tokens, 2 per block
  stack_kernel<<<2048, NTHREADS, LDS_BYTES, stream>>>(
      x, /*mode=*/0, /*S=*/62, blob_c,
      (const float*)d_in[5], (const float*)d_in[6], (const float*)d_in[7],
      (const float*)d_in[8], (const float*)d_in[9], (const float*)d_in[11],
      (const float*)d_in[13], stageA);

  // Stage B: temporal attention, 64 sequences of 64 tokens, 2 per block -> d_out
  stack_kernel<<<32, NTHREADS, LDS_BYTES, stream>>>(
      stageA, /*mode=*/1, /*S=*/64, blob_t,
      (const float*)d_in[18], (const float*)d_in[19], (const float*)d_in[20],
      (const float*)d_in[21], (const float*)d_in[22], (const float*)d_in[24],
      (const float*)d_in[26], (float*)d_out);
}
